// AttentionBlock_63677185130724
// MI455X (gfx1250) — matrix operand order repaired and checked by reading, not compiled, not run on hardware
//
#include <hip/hip_runtime.h>

typedef __attribute__((ext_vector_type(16))) _Float16 v16h;
typedef __attribute__((ext_vector_type(8)))  float    v8f;
typedef _Float16 half_t;

#define BATCH 16
#define CCH   512
#define SEQ   1024
#define NHEAD 4
#define HDIM  128

// ---------------------------------------------------------------------------
// GroupNorm(num_groups=1): per-batch mean / rstd over C*H*W = 512*1024 elems
// ---------------------------------------------------------------------------
__global__ __launch_bounds__(256)
void gn_stats_kernel(const float* __restrict__ x, float* __restrict__ stats) {
    const int b = blockIdx.x;
    const float* xb = x + (size_t)b * (CCH * SEQ);
    float s = 0.f, ss = 0.f;
    for (int i = threadIdx.x; i < CCH * SEQ; i += 256) {
        float v = xb[i];
        s += v; ss += v * v;
    }
    __shared__ float rs[256], rq[256];
    rs[threadIdx.x] = s; rq[threadIdx.x] = ss;
    __syncthreads();
    for (int st = 128; st > 0; st >>= 1) {
        if (threadIdx.x < st) {
            rs[threadIdx.x] += rs[threadIdx.x + st];
            rq[threadIdx.x] += rq[threadIdx.x + st];
        }
        __syncthreads();
    }
    if (threadIdx.x == 0) {
        const float inv = 1.0f / (float)(CCH * SEQ);
        float mean = rs[0] * inv;
        float var  = rq[0] * inv - mean * mean;
        stats[b * 2 + 0] = mean;
        stats[b * 2 + 1] = rsqrtf(var + 1e-5f);
    }
}

// normalize, apply gamma/beta, transpose (B,C,S) -> (B,S,C), convert to f16
__global__ __launch_bounds__(256)
void gn_apply_kernel(const float* __restrict__ x, const float* __restrict__ stats,
                     const float* __restrict__ gw, const float* __restrict__ gb,
                     half_t* __restrict__ seq) {
    size_t i = (size_t)blockIdx.x * 256 + threadIdx.x;   // < 16*512*1024 = 8M
    int b = (int)(i >> 19);
    int c = (int)((i >> 10) & (CCH - 1));
    int s = (int)(i & (SEQ - 1));
    float mean = stats[b * 2 + 0];
    float rstd = stats[b * 2 + 1];
    float v = (x[i] - mean) * rstd * gw[c] + gb[c];
    seq[((size_t)b << 19) + ((size_t)s << 9) + c] = (half_t)v;
}

__global__ __launch_bounds__(256)
void f32_to_f16_kernel(const float* __restrict__ a, half_t* __restrict__ o, int n) {
    int i = blockIdx.x * 256 + threadIdx.x;
    if (i < n) o[i] = (half_t)a[i];
}

// ---------------------------------------------------------------------------
// WMMA GEMM:  Y[s][o] = sum_c A[s][c] * W[o][c] + bias[o]
//   mode 0: store f16, (B,S,C) row-major              (Q / K projections)
//   mode 2: store f16 transposed per head: Vt[b][h][d][s]   (V projection)
//   mode 1: f32 out + residual, transposed to (B,C,S)   (output projection)
// grid (S/16, C/128, B), 256 threads = 8 waves; block stages 16 rows of A in
// LDS (16 KB), each wave owns one 16x16 output tile and runs 16 WMMAs (K=512).
// ---------------------------------------------------------------------------
__global__ __launch_bounds__(256)
void gemm_wT_kernel(const half_t* __restrict__ A, const half_t* __restrict__ W,
                    const float* __restrict__ bias,
                    half_t* __restrict__ outH,
                    const float* __restrict__ resX, float* __restrict__ outF,
                    int mode) {
    __shared__ half_t As[16][520];                 // +8 halves pad, 16B aligned rows
    const int b    = blockIdx.z;
    const int m0   = blockIdx.x * 16;
    const int wave = threadIdx.x >> 5;
    const int lane = threadIdx.x & 31;
    const int n0   = (blockIdx.y * 8 + wave) * 16;

    const half_t* Ab = A + ((size_t)b * SEQ + m0) * CCH;
    {   // cooperative stage: each thread copies 32 contiguous halves (64 B)
        int r  = threadIdx.x >> 4;
        int cc = (threadIdx.x & 15) * 32;
        const uint4* src = (const uint4*)(Ab + (size_t)r * CCH + cc);
        uint4* dst = (uint4*)(&As[r][cc]);
        dst[0] = src[0]; dst[1] = src[1]; dst[2] = src[2]; dst[3] = src[3];
    }
    __syncthreads();

    const int ln = lane & 15;       // A: row m / B: col n / D: col n
    const int g  = lane >> 4;       // half-wave group
    v8f acc = {};

#pragma unroll
    for (int kk = 0; kk < CCH; kk += 32) {
        v16h af, bf;
        ((uint4*)&af)[0] = *(const uint4*)&As[ln][kk + g * 16];
        ((uint4*)&af)[1] = *(const uint4*)&As[ln][kk + g * 16 + 8];
        const half_t* wr = W + (size_t)(n0 + ln) * CCH + kk + g * 16;
        ((uint4*)&bf)[0] = ((const uint4*)wr)[0];
        ((uint4*)&bf)[1] = ((const uint4*)wr)[1];
        acc = __builtin_amdgcn_wmma_f32_16x16x32_f16(false, af, false, bf,
                                                     (short)0, acc, false, false);
    }

    const float bv = bias[n0 + ln];
#pragma unroll
    for (int r = 0; r < 8; ++r) acc[r] += bv;

    if (mode == 0) {                         // (B,S,C) f16
#pragma unroll
        for (int r = 0; r < 8; ++r) {
            int srow = m0 + r + g * 8;
            outH[((size_t)b * SEQ + srow) * CCH + n0 + ln] = (half_t)acc[r];
        }
    } else if (mode == 2) {                  // Vt[b][h][d][s] f16
        int cch = n0 + ln;
        int h = cch >> 7, d = cch & (HDIM - 1);
        half_t* vt = outH + (((size_t)(b * NHEAD + h) * HDIM + d) << 10) + m0 + g * 8;
#pragma unroll
        for (int r = 0; r < 8; ++r) vt[r] = (half_t)acc[r];
    } else {                                 // (B,C,S) f32 + residual
        int cch = n0 + ln;
        size_t base = ((size_t)b * CCH + cch) * SEQ + m0 + g * 8;
#pragma unroll
        for (int r = 0; r < 8; ++r) outF[base + r] = acc[r] + resX[base + r];
    }
}

// ---------------------------------------------------------------------------
// Flash attention: wave owns 16 query rows of one (batch, head).
// 32 keys / iter: 8 score WMMAs, online softmax, P via private LDS tile,
// 8 P*V WMMAs (Vt is d-major so B-fragments are contiguous 16B loads).
// ---------------------------------------------------------------------------
__global__ __launch_bounds__(256)
void attention_kernel(const half_t* __restrict__ Q, const half_t* __restrict__ K,
                      const half_t* __restrict__ Vt, half_t* __restrict__ O) {
    __shared__ half_t Pl[8][16 * 32];              // 1 KB per wave
    const int wave = threadIdx.x >> 5;
    const int lane = threadIdx.x & 31;
    const int task = blockIdx.x * 8 + wave;        // 4096 tasks total
    const int qt = task & 63;
    const int h  = (task >> 6) & 3;
    const int b  = task >> 8;
    const int q0 = qt * 16;

    const half_t* Qb = Q  + ((size_t)b << 19) + (h << 7);
    const half_t* Kb = K  + ((size_t)b << 19) + (h << 7);
    const half_t* Vb = Vt + ((size_t)(b * NHEAD + h) << 17);
    half_t* Pw = &Pl[wave][0];

    const int ln = lane & 15;
    const int g  = lane >> 4;

    v16h qf[4];
#pragma unroll
    for (int t = 0; t < 4; ++t) {
        const half_t* qr = Qb + (size_t)(q0 + ln) * CCH + t * 32;
        ((uint4*)&qf[t])[0] = *(const uint4*)(qr + g * 16);
        ((uint4*)&qf[t])[1] = *(const uint4*)(qr + g * 16 + 8);
    }

    float mi[8], li[8];
    v8f oacc[8];
#pragma unroll
    for (int r = 0; r < 8; ++r) { mi[r] = -1e30f; li[r] = 0.f; }
#pragma unroll
    for (int t = 0; t < 8; ++t)
#pragma unroll
        for (int r = 0; r < 8; ++r) oacc[t][r] = 0.f;

    const float sc = 0.08838834764831845f;         // 1/sqrt(128)

    for (int j0 = 0; j0 < SEQ; j0 += 32) {
        // ---- scores: two 16x16 tiles over [j0, j0+32)
        v8f s01[2];
#pragma unroll
        for (int jj = 0; jj < 2; ++jj) {
            v8f s = {};
#pragma unroll
            for (int t = 0; t < 4; ++t) {
                v16h kf;
                const half_t* kr = Kb + (size_t)(j0 + jj * 16 + ln) * CCH + t * 32 + g * 16;
                ((uint4*)&kf)[0] = ((const uint4*)kr)[0];
                ((uint4*)&kf)[1] = ((const uint4*)kr)[1];
                s = __builtin_amdgcn_wmma_f32_16x16x32_f16(false, qf[t], false, kf,
                                                           (short)0, s, false, false);
            }
            s01[jj] = s;
        }
        __builtin_prefetch(Kb + (size_t)(j0 + 32 + ln) * CCH, 0, 1);

        // ---- online softmax (rows live across 16-lane halves)
        float alpha[8];
#pragma unroll
        for (int r = 0; r < 8; ++r) {
            float a = s01[0][r] * sc, c = s01[1][r] * sc;
            float mx = fmaxf(a, c);
#pragma unroll
            for (int off = 1; off < 16; off <<= 1)
                mx = fmaxf(mx, __shfl_xor(mx, off, 32));
            float mnew = fmaxf(mi[r], mx);
            alpha[r] = __expf(mi[r] - mnew);
            float p0 = __expf(a - mnew);
            float p1 = __expf(c - mnew);
            s01[0][r] = p0; s01[1][r] = p1;
            float rs = p0 + p1;
#pragma unroll
            for (int off = 1; off < 16; off <<= 1)
                rs += __shfl_xor(rs, off, 32);
            li[r] = li[r] * alpha[r] + rs;
            mi[r] = mnew;
        }
#pragma unroll
        for (int t = 0; t < 8; ++t)
#pragma unroll
            for (int r = 0; r < 8; ++r) oacc[t][r] *= alpha[r];

        // ---- P: D-layout -> LDS -> A-layout (wave-private tile)
#pragma unroll
        for (int r = 0; r < 8; ++r) {
            int m = r + g * 8;
            Pw[m * 32 + ln]      = (half_t)s01[0][r];
            Pw[m * 32 + 16 + ln] = (half_t)s01[1][r];
        }
        __builtin_amdgcn_sched_barrier(0);
        asm volatile("s_wait_dscnt 0x0" ::: "memory");   // in-wave DS RAW ordering
        v16h pf;
        ((uint4*)&pf)[0] = *(const uint4*)&Pw[ln * 32 + g * 16];
        ((uint4*)&pf)[1] = *(const uint4*)&Pw[ln * 32 + g * 16 + 8];
        asm volatile("" ::: "memory");                   // keep next-iter stores below loads

        // ---- P * V  (Vt d-major: B-frag K contiguous)
#pragma unroll
        for (int t = 0; t < 8; ++t) {
            v16h vf;
            const half_t* vr = Vb + (size_t)(t * 16 + ln) * SEQ + j0 + g * 16;
            ((uint4*)&vf)[0] = ((const uint4*)vr)[0];
            ((uint4*)&vf)[1] = ((const uint4*)vr)[1];
            oacc[t] = __builtin_amdgcn_wmma_f32_16x16x32_f16(false, pf, false, vf,
                                                             (short)0, oacc[t], false, false);
        }
    }

    // ---- epilogue: normalize and store (B,S,C) f16
#pragma unroll
    for (int r = 0; r < 8; ++r) li[r] = 1.0f / li[r];
#pragma unroll
    for (int t = 0; t < 8; ++t)
#pragma unroll
        for (int r = 0; r < 8; ++r) {
            int srow = q0 + r + g * 8;
            int d = h * HDIM + t * 16 + ln;
            O[((size_t)b * SEQ + srow) * CCH + d] = (half_t)(oacc[t][r] * li[r]);
        }
}

// ---------------------------------------------------------------------------
extern "C" void kernel_launch(void* const* d_in, const int* in_sizes, int n_in,
                              void* d_out, int out_size, void* d_ws, size_t ws_size,
                              hipStream_t stream) {
    const float* x  = (const float*)d_in[0];
    const float* gw = (const float*)d_in[1];
    const float* gb = (const float*)d_in[2];
    const float* wq = (const float*)d_in[3];
    const float* bq = (const float*)d_in[4];
    const float* wk = (const float*)d_in[5];
    const float* bk = (const float*)d_in[6];
    const float* wv = (const float*)d_in[7];
    const float* bv = (const float*)d_in[8];
    const float* wo = (const float*)d_in[9];
    const float* bo = (const float*)d_in[10];
    float* out = (float*)d_out;

    char* ws = (char*)d_ws;
    const size_t actElems = (size_t)BATCH * SEQ * CCH;          // 8M
    size_t off = 0;
    float*  stats = (float*)(ws + off);  off += 256;
    half_t* seq = (half_t*)(ws + off);   off += actElems * 2;   // (B,S,C)
    half_t* qh  = (half_t*)(ws + off);   off += actElems * 2;
    half_t* kh  = (half_t*)(ws + off);   off += actElems * 2;
    half_t* vt  = (half_t*)(ws + off);   off += actElems * 2;   // (B,H,D,S)
    half_t* ao  = (half_t*)(ws + off);   off += actElems * 2;   // attn out (B,S,C)
    half_t* wqh = (half_t*)(ws + off);   off += (size_t)CCH * CCH * 2;
    half_t* wkh = (half_t*)(ws + off);   off += (size_t)CCH * CCH * 2;
    half_t* wvh = (half_t*)(ws + off);   off += (size_t)CCH * CCH * 2;
    half_t* woh = (half_t*)(ws + off);   off += (size_t)CCH * CCH * 2;

    gn_stats_kernel<<<BATCH, 256, 0, stream>>>(x, stats);
    gn_apply_kernel<<<(unsigned)(actElems / 256), 256, 0, stream>>>(x, stats, gw, gb, seq);

    const int wn = CCH * CCH;
    f32_to_f16_kernel<<<(wn + 255) / 256, 256, 0, stream>>>(wq, wqh, wn);
    f32_to_f16_kernel<<<(wn + 255) / 256, 256, 0, stream>>>(wk, wkh, wn);
    f32_to_f16_kernel<<<(wn + 255) / 256, 256, 0, stream>>>(wv, wvh, wn);
    f32_to_f16_kernel<<<(wn + 255) / 256, 256, 0, stream>>>(wo, woh, wn);

    dim3 ggrid(SEQ / 16, CCH / 128, BATCH);   // (64, 4, 16)
    gemm_wT_kernel<<<ggrid, 256, 0, stream>>>(seq, wqh, bq, qh, nullptr, nullptr, 0);
    gemm_wT_kernel<<<ggrid, 256, 0, stream>>>(seq, wkh, bk, kh, nullptr, nullptr, 0);
    gemm_wT_kernel<<<ggrid, 256, 0, stream>>>(seq, wvh, bv, vt, nullptr, nullptr, 2);

    attention_kernel<<<512, 256, 0, stream>>>(qh, kh, vt, ao);

    gemm_wT_kernel<<<ggrid, 256, 0, stream>>>(ao, woh, bo, nullptr, x, out, 1);
}
